// MultiHeadAttention_54606214201531
// MI455X (gfx1250) — matrix operand order repaired and checked by reading, not compiled, not run on hardware
//
#include <hip/hip_runtime.h>

typedef __attribute__((ext_vector_type(16))) __bf16      v16bf;
typedef __attribute__((ext_vector_type(8)))  float       v8f;
typedef __attribute__((ext_vector_type(4)))  unsigned int u32x4;

#define AB  4      // batch
#define AS  2048   // sequence
#define AD  1024   // model dim
#define AH  16     // heads
#define ADH 64     // head dim

// ---------------------------------------------------------------- helpers ---

__device__ __forceinline__ unsigned short f2bf(float f) {
    union { float f; unsigned int u; } c; c.f = f;
    unsigned int u = c.u;
    return (unsigned short)((u + 0x7FFFu + ((u >> 16) & 1u)) >> 16);  // RNE
}

union BFrag {
    v16bf v;
    u32x4 q[2];
};

// A-fragment (16x32 bf16): lane holds row (lane&15); element j -> K =
// j + (lane<16?0:16).  rowptr must already include the
// per-half +16 element offset; chunk1 lives at +8 elements.
__device__ __forceinline__ BFrag load_a_frag(const unsigned short* rowptr) {
    BFrag f;
    f.q[0] = *(const u32x4*)(rowptr);
    f.q[1] = *(const u32x4*)(rowptr + 8);
    return f;
}

// B-fragment (32x16 bf16): lane holds column (lane&15); element j -> K =
// j + (lane<16?0:16).  p must already include the per-half +16 offset:
// 16 contiguous bf16 (32 bytes).
__device__ __forceinline__ BFrag load_b_frag(const unsigned short* p) {
    BFrag f;
    f.q[0] = *(const u32x4*)(p);
    f.q[1] = *(const u32x4*)(p + 8);
    return f;
}

// ------------------------------------------------------------- fp32->bf16 ---

__global__ void cvt_f32_bf16(const float* __restrict__ in,
                             unsigned short* __restrict__ out, int n) {
    int i = blockIdx.x * blockDim.x + threadIdx.x;
    int stride = gridDim.x * blockDim.x;
    for (; i < n; i += stride) out[i] = f2bf(in[i]);
}

// ------------------------------------------------------------------- GEMM ---
// C[m,n] = sum_k A[m,k] * W[n,k]   (A: MxK bf16 row-major, W: NxK bf16)
// One wave computes a 32(M) x 64(N) tile: 2 M-subtiles x 4 N-subtiles, so
// every B-fragment load feeds two WMMAs (24 B of traffic per WMMA).
// mode 0: f32 row-major [M,N] out
// mode 1: bf16 head-split out  [B,H,S,ADH]   (for Q and K)
// mode 2: bf16 transposed out  [B,H,ADH,S]   (for V)

__device__ __forceinline__ void gemm_epilogue(v8f acc[4], int mbase, int ncol,
                                              int N, int mode, void* outp) {
    if (mode == 0) {
        float* out = (float*)outp;
#pragma unroll
        for (int r = 0; r < 8; ++r) {
            float* o = out + (size_t)(mbase + r) * N + ncol;
#pragma unroll
            for (int nt = 0; nt < 4; ++nt) o[nt * 16] = acc[nt][r];
        }
    } else if (mode == 1) {
        unsigned short* out = (unsigned short*)outp;
#pragma unroll
        for (int r = 0; r < 8; ++r) {
            int m = mbase + r;
            int b_ = m / AS, s = m % AS;
#pragma unroll
            for (int nt = 0; nt < 4; ++nt) {
                int n = ncol + nt * 16;
                int h = n >> 6, d = n & 63;
                out[((size_t)(b_ * AH + h) * AS + s) * ADH + d] = f2bf(acc[nt][r]);
            }
        }
    } else {  // mode 2: V transposed [B,H,ADH,S]
        unsigned short* out = (unsigned short*)outp;
#pragma unroll
        for (int r = 0; r < 8; ++r) {
            int m = mbase + r;
            int b_ = m / AS, s = m % AS;
#pragma unroll
            for (int nt = 0; nt < 4; ++nt) {
                int n = ncol + nt * 16;
                int h = n >> 6, d = n & 63;
                out[((size_t)(b_ * AH + h) * ADH + d) * AS + s] = f2bf(acc[nt][r]);
            }
        }
    }
}

__global__ void __launch_bounds__(128)
gemm_bf16_wmma(const unsigned short* __restrict__ A,
               const unsigned short* __restrict__ W,
               void* __restrict__ outp,
               int M, int N, int K, int mode)
{
    const int lane  = threadIdx.x & 31;
    const int wid   = threadIdx.x >> 5;
    const int tiles_n = N >> 6;
    const int gid   = blockIdx.x * 4 + wid;
    const int mtile = gid / tiles_n;               // 32-row tiles
    const int ntile = gid % tiles_n;               // 64-col tiles
    if (mtile * 32 >= M) return;                   // wave-uniform

    const int lhalf = lane >> 4;
    const int l15   = lane & 15;

    const unsigned short* arow0 = A + (size_t)(mtile * 32 + l15) * K + lhalf * 16;
    const unsigned short* arow1 = arow0 + (size_t)16 * K;
    const unsigned short* wrow  = W + (size_t)(ntile * 64 + l15) * K + lhalf * 16;

    v8f acc0[4] = {};
    v8f acc1[4] = {};

    for (int k0 = 0; k0 < K; k0 += 32) {
        // prefetch two K-steps ahead, WGP scope (pull into all cache levels)
        __builtin_prefetch(arow0 + k0 + 64, 0, 3);
        __builtin_prefetch(arow1 + k0 + 64, 0, 3);
        __builtin_prefetch(wrow + k0 + 64, 0, 3);

        BFrag a0 = load_a_frag(arow0 + k0);
        BFrag a1 = load_a_frag(arow1 + k0);
        BFrag b[4];
#pragma unroll
        for (int nt = 0; nt < 4; ++nt)
            b[nt] = load_b_frag(wrow + (size_t)(nt * 16) * K + k0);
#pragma unroll
        for (int nt = 0; nt < 4; ++nt) {
            acc0[nt] = __builtin_amdgcn_wmma_f32_16x16x32_bf16(
                false, a0.v, false, b[nt].v, (short)0, acc0[nt], false, false);
            acc1[nt] = __builtin_amdgcn_wmma_f32_16x16x32_bf16(
                false, a1.v, false, b[nt].v, (short)0, acc1[nt], false, false);
        }
    }

    const int ncol = ntile * 64 + l15;
    gemm_epilogue(acc0, mtile * 32 + lhalf * 8,      ncol, N, mode, outp);
    gemm_epilogue(acc1, mtile * 32 + 16 + lhalf * 8, ncol, N, mode, outp);
}

// -------------------------------------------------------- flash attention ---
// One wave: 16 queries x dh=64 output; streams 32-key blocks.
// Qh, Kh: [B,H,S,ADH] bf16.  Vt: [B,H,ADH,S] bf16.  Obuf: [B*S, D] bf16.

__global__ void __launch_bounds__(128)
attn_wmma(const unsigned short* __restrict__ Qh,
          const unsigned short* __restrict__ Kh,
          const unsigned short* __restrict__ Vt,
          unsigned short* __restrict__ Obuf)
{
    __shared__ __align__(16) unsigned short pl[4][16 * 32];  // P-tile staging

    const int lane = threadIdx.x & 31;
    const int wid  = threadIdx.x >> 5;
    const int gid  = blockIdx.x * 4 + wid;
    const int qt   = gid & 127;     // S/16 = 128 query tiles
    const int bh   = gid >> 7;      // 0..63
    const int b_   = bh >> 4, h = bh & 15;

    const int lhalf = lane >> 4;
    const int l15   = lane & 15;

    const unsigned short* Qbase = Qh + (size_t)bh * AS * ADH;
    const unsigned short* Kbase = Kh + (size_t)bh * AS * ADH;
    const unsigned short* Vbase = Vt + (size_t)bh * ADH * AS;

    // Q fragments, loaded once (d 0..31 and 32..63)
    const unsigned short* qrow = Qbase + (size_t)(qt * 16 + l15) * ADH + lhalf * 16;
    BFrag qa0 = load_a_frag(qrow);
    BFrag qa1 = load_a_frag(qrow + 32);

    float rm[8], rs[8];
#pragma unroll
    for (int r = 0; r < 8; ++r) { rm[r] = -1.0e30f; rs[r] = 0.0f; }
    v8f o[4] = {};

    unsigned short* plw = &pl[wid][0];
    const float scale = 0.125f;     // 1/sqrt(64)

    for (int kb = 0; kb < AS; kb += 32) {
        // ---- scores: S[16 x 32] = Q Kt, two 16x16 tiles, K-dim = 64 (2 steps)
        const unsigned short* krow0 = Kbase + (size_t)(kb + l15) * ADH + lhalf * 16;
        const unsigned short* krow1 = krow0 + 16 * ADH;
        BFrag kf00 = load_b_frag(krow0);        // keys kb..kb+15,  d 0..31
        BFrag kf01 = load_b_frag(krow0 + 32);   //                  d 32..63
        BFrag kf10 = load_b_frag(krow1);        // keys kb+16..31
        BFrag kf11 = load_b_frag(krow1 + 32);

        v8f s0 = {}, s1 = {};
        s0 = __builtin_amdgcn_wmma_f32_16x16x32_bf16(false, qa0.v, false, kf00.v, (short)0, s0, false, false);
        s0 = __builtin_amdgcn_wmma_f32_16x16x32_bf16(false, qa1.v, false, kf01.v, (short)0, s0, false, false);
        s1 = __builtin_amdgcn_wmma_f32_16x16x32_bf16(false, qa0.v, false, kf10.v, (short)0, s1, false, false);
        s1 = __builtin_amdgcn_wmma_f32_16x16x32_bf16(false, qa1.v, false, kf11.v, (short)0, s1, false, false);

        // ---- online softmax, row-wise (row = 16-lane group per C-layout)
#pragma unroll
        for (int r = 0; r < 8; ++r) {
            float v0 = s0[r] * scale, v1 = s1[r] * scale;
            float mx = fmaxf(v0, v1);
            mx = fmaxf(mx, __shfl_xor(mx, 1, 32));
            mx = fmaxf(mx, __shfl_xor(mx, 2, 32));
            mx = fmaxf(mx, __shfl_xor(mx, 4, 32));
            mx = fmaxf(mx, __shfl_xor(mx, 8, 32));
            float nm   = fmaxf(rm[r], mx);
            float corr = __expf(rm[r] - nm);
            float p0   = __expf(v0 - nm);
            float p1   = __expf(v1 - nm);
            float ps = p0 + p1;
            ps += __shfl_xor(ps, 1, 32);
            ps += __shfl_xor(ps, 2, 32);
            ps += __shfl_xor(ps, 4, 32);
            ps += __shfl_xor(ps, 8, 32);
            rs[r] = rs[r] * corr + ps;
            rm[r] = nm;
#pragma unroll
            for (int nt = 0; nt < 4; ++nt) o[nt][r] *= corr;

            int m = r + lhalf * 8;             // C-layout row for this VGPR
            plw[m * 32 + l15]      = f2bf(p0);
            plw[m * 32 + 16 + l15] = f2bf(p1);
        }
        asm volatile("s_wait_dscnt 0" ::: "memory");

        // ---- P (C-layout in LDS) re-read as A-fragment
        BFrag pa = load_a_frag(plw + l15 * 32 + lhalf * 16);

        // ---- O += P V  (K-dim = 32 keys; B operand from transposed V)
        const unsigned short* vcol = Vbase + (size_t)l15 * AS + kb + lhalf * 16;
#pragma unroll
        for (int nt = 0; nt < 4; ++nt) {
            BFrag vf = load_b_frag(vcol + (size_t)(nt * 16) * AS);
            o[nt] = __builtin_amdgcn_wmma_f32_16x16x32_bf16(
                false, pa.v, false, vf.v, (short)0, o[nt], false, false);
        }
    }

    // ---- normalize and write concat-head bf16 output [B*S, D]
#pragma unroll
    for (int r = 0; r < 8; ++r) {
        float inv = 1.0f / rs[r];
        int s = qt * 16 + r + lhalf * 8;
        unsigned short* op = Obuf + ((size_t)(b_ * AS + s)) * AD + h * ADH + l15;
#pragma unroll
        for (int nt = 0; nt < 4; ++nt) op[nt * 16] = f2bf(o[nt][r] * inv);
    }
}

// ---------------------------------------------------------------- launch ----

extern "C" void kernel_launch(void* const* d_in, const int* in_sizes, int n_in,
                              void* d_out, int out_size, void* d_ws, size_t ws_size,
                              hipStream_t stream) {
    const float* x  = (const float*)d_in[0];
    const float* Wq = (const float*)d_in[1];
    const float* Wk = (const float*)d_in[2];
    const float* Wv = (const float*)d_in[3];
    const float* Wo = (const float*)d_in[4];

    const size_t nX = (size_t)AB * AS * AD;   // 8,388,608
    const size_t nW = (size_t)AD * AD;        // 1,048,576

    char* ws = (char*)d_ws;
    unsigned short* xb   = (unsigned short*)ws; ws += nX * 2;
    unsigned short* Wqb  = (unsigned short*)ws; ws += nW * 2;
    unsigned short* Wkb  = (unsigned short*)ws; ws += nW * 2;
    unsigned short* Wvb  = (unsigned short*)ws; ws += nW * 2;
    unsigned short* Wob  = (unsigned short*)ws; ws += nW * 2;
    unsigned short* Qh   = (unsigned short*)ws; ws += nX * 2;
    unsigned short* Kh   = (unsigned short*)ws; ws += nX * 2;
    unsigned short* Vt   = (unsigned short*)ws; ws += nX * 2;
    unsigned short* Obuf = (unsigned short*)ws; ws += nX * 2;

    cvt_f32_bf16<<<4096, 256, 0, stream>>>(x,  xb,  (int)nX);
    cvt_f32_bf16<<<1024, 256, 0, stream>>>(Wq, Wqb, (int)nW);
    cvt_f32_bf16<<<1024, 256, 0, stream>>>(Wk, Wkb, (int)nW);
    cvt_f32_bf16<<<1024, 256, 0, stream>>>(Wv, Wvb, (int)nW);
    cvt_f32_bf16<<<1024, 256, 0, stream>>>(Wo, Wob, (int)nW);

    const int M = AB * AS;                       // 8192
    const int wave_tiles = (M / 32) * (AD / 64); // 4096
    const int gemm_blocks = wave_tiles / 4;      // 1024

    gemm_bf16_wmma<<<gemm_blocks, 128, 0, stream>>>(xb, Wqb, Qh, M, AD, AD, 1);
    gemm_bf16_wmma<<<gemm_blocks, 128, 0, stream>>>(xb, Wkb, Kh, M, AD, AD, 1);
    gemm_bf16_wmma<<<gemm_blocks, 128, 0, stream>>>(xb, Wvb, Vt, M, AD, AD, 2);

    // (B*H) * (S/16) waves = 8192 -> 2048 blocks of 4 waves
    attn_wmma<<<2048, 128, 0, stream>>>(Qh, Kh, Vt, Obuf);

    gemm_bf16_wmma<<<gemm_blocks, 128, 0, stream>>>(Obuf, Wob, d_out, M, AD, AD, 0);
}
